// Attention_4088808865947
// MI455X (gfx1250) — matrix operand order repaired and checked by reading, compiled, not run on hardware
//
// CDNA5 / gfx1250 (MI455X) attention block:
//   split(x,wqkv) -> QKV GEMM (bf16x3 WMMA) -> RMSNorm+RoPE (emit pre-split
//   bf16 q/k, transposed pre-split v) -> causal GQA flash attention (bf16x3
//   WMMA for QK^T and P.V) -> split(y,wo) -> output GEMM (bf16x3 WMMA).
//
// The f32->bf16 hi/lo error-compensation split is hoisted OUT of all WMMA
// inner loops into one-shot elementwise passes.  The GEMM hot loop is
// 16x global_load_b128 + 12x v_wmma per K=32 step (near-zero VALU).
//
// Roofline: ~155 GFLOP, ~600 MB total HBM traffic (incl. split passes)
// -> ~26 us memory floor @ 23.3 TB/s; bf16x3 WMMA issue ~360 GF-equivalent.

#include <hip/hip_runtime.h>

#define S_LEN  2048
#define DMODEL 2048
#define NHEADS 16
#define NKVH   4
#define HDIM   128

typedef __attribute__((ext_vector_type(16))) __bf16 v16bf;
typedef __attribute__((ext_vector_type(8)))  __bf16 v8bf;
typedef __attribute__((ext_vector_type(4)))  __bf16 v4bf;
typedef __attribute__((ext_vector_type(8)))  float  v8f;

// ---- bf16 helpers (bit-exact RNE, no reliance on native __bf16 arithmetic) ----
__device__ __forceinline__ __bf16 f2bf(float f) {
  unsigned int u = __builtin_bit_cast(unsigned int, f);
  unsigned int r = u + 0x7FFFu + ((u >> 16) & 1u);
  return __builtin_bit_cast(__bf16, (unsigned short)(r >> 16));
}
__device__ __forceinline__ float bf2f(__bf16 h) {
  unsigned int u = ((unsigned int)__builtin_bit_cast(unsigned short, h)) << 16;
  return __builtin_bit_cast(float, u);
}
struct bfpair { __bf16 h; __bf16 l; };
__device__ __forceinline__ bfpair split1(float x) {
  bfpair p;
  p.h = f2bf(x);
  p.l = f2bf(x - bf2f(p.h));
  return p;
}

__device__ __forceinline__ v8f wmma_bf16(v16bf a, v16bf b, v8f c) {
  // 8 args: (neg_a, A, neg_b, B, c_mod, C, reuse_a, reuse_b)
  return __builtin_amdgcn_wmma_f32_16x16x32_bf16(false, a, false, b, (short)0, c,
                                                 false, false);
}
__device__ __forceinline__ v16bf cat8(v8bf a, v8bf b) {
  return __builtin_shufflevector(a, b, 0, 1, 2, 3, 4, 5, 6, 7,
                                 8, 9, 10, 11, 12, 13, 14, 15);
}
__device__ __forceinline__ v8bf ld8(const __bf16* p) { return *(const v8bf*)p; }

// A-fragment 16(M)x32(K): lane m=lane&15;
// elems 0..15 = K g..g+15, g = 16*(lane>>4).  rowp = base + m*ld + k0.
__device__ __forceinline__ v16bf load_a_frag(const __bf16* rowp, int lane) {
  const int g = (lane >> 4) * 16;
  return cat8(ld8(rowp + g), ld8(rowp + g + 8));
}
// B-fragment 32(K)x16(N) from row-major [N,K]: lane n=lane&15;
// elems 0..15 = K g..g+15, g = 16*(lane>>4).  rowp = base + n*ld + k0.
__device__ __forceinline__ v16bf load_b_frag(const __bf16* rowp, int lane) {
  const int g = (lane >> 4) * 16;
  return cat8(ld8(rowp + g), ld8(rowp + g + 8));
}

// ==================  elementwise f32 -> bf16 hi/lo split  ==================
__global__ __launch_bounds__(256) void split_f32_kernel(
    const float* __restrict__ in, __bf16* __restrict__ hi,
    __bf16* __restrict__ lo, int n4) {
  const int i = blockIdx.x * 256 + threadIdx.x;
  if (i >= n4) return;
  const float4 v = ((const float4*)in)[i];
  v4bf h, l;
  bfpair p;
  p = split1(v.x); h[0] = p.h; l[0] = p.l;
  p = split1(v.y); h[1] = p.h; l[1] = p.l;
  p = split1(v.z); h[2] = p.h; l[2] = p.l;
  p = split1(v.w); h[3] = p.h; l[3] = p.l;
  ((v4bf*)hi)[i] = h;
  ((v4bf*)lo)[i] = l;
}

// ==============  NT GEMM on pre-split bf16x3 operands, C fp32  =============
// C[M,N] = A[M,K] * B[N,K]^T.  256 threads = 8 waves -> 64(M)x128(N) block
// tile; each wave owns 32x32 (4 accumulators, 12 WMMAs / K=32 step).
__global__ __launch_bounds__(256) void gemm_nt_bf16x3(
    const __bf16* __restrict__ Ahi, const __bf16* __restrict__ Alo,
    const __bf16* __restrict__ Bhi, const __bf16* __restrict__ Blo,
    float* __restrict__ C, int M, int N, int K) {
  (void)M;
  const int lane = threadIdx.x & 31;
  const int wave = threadIdx.x >> 5;
  const int m0 = blockIdx.y * 64 + (wave >> 2) * 32;
  const int n0 = blockIdx.x * 128 + (wave & 3) * 32;

  const size_t ar0 = (size_t)(m0 + (lane & 15)) * K;
  const size_t ar1 = ar0 + (size_t)16 * K;
  const size_t br0 = (size_t)(n0 + (lane & 15)) * K;
  const size_t br1 = br0 + (size_t)16 * K;

  v8f c00, c01, c10, c11;
#pragma unroll
  for (int r = 0; r < 8; r++) { c00[r] = 0.f; c01[r] = 0.f; c10[r] = 0.f; c11[r] = 0.f; }

  for (int k = 0; k < K; k += 32) {
    __builtin_prefetch(Ahi + ar0 + k + 128, 0, 1);  // global_prefetch_b8
    __builtin_prefetch(Bhi + br0 + k + 128, 0, 1);
    const v16bf a0h = load_a_frag(Ahi + ar0 + k, lane);
    const v16bf a0l = load_a_frag(Alo + ar0 + k, lane);
    const v16bf a1h = load_a_frag(Ahi + ar1 + k, lane);
    const v16bf a1l = load_a_frag(Alo + ar1 + k, lane);
    const v16bf b0h = load_b_frag(Bhi + br0 + k, lane);
    const v16bf b0l = load_b_frag(Blo + br0 + k, lane);
    const v16bf b1h = load_b_frag(Bhi + br1 + k, lane);
    const v16bf b1l = load_b_frag(Blo + br1 + k, lane);
    c00 = wmma_bf16(a0h, b0h, c00); c00 = wmma_bf16(a0l, b0h, c00); c00 = wmma_bf16(a0h, b0l, c00);
    c01 = wmma_bf16(a0h, b1h, c01); c01 = wmma_bf16(a0l, b1h, c01); c01 = wmma_bf16(a0h, b1l, c01);
    c10 = wmma_bf16(a1h, b0h, c10); c10 = wmma_bf16(a1l, b0h, c10); c10 = wmma_bf16(a1h, b0l, c10);
    c11 = wmma_bf16(a1h, b1h, c11); c11 = wmma_bf16(a1l, b1h, c11); c11 = wmma_bf16(a1h, b1l, c11);
  }

  // C/D layout: element (m,n): lane = n + 16*(m>>3), vgpr = m&7.
  const int cn = lane & 15;
  const int cm = (lane >> 4) * 8;
  float* Cp = C + (size_t)(m0 + cm) * N + n0 + cn;
#pragma unroll
  for (int r = 0; r < 8; r++) {
    Cp[(size_t)r * N] = c00[r];
    Cp[(size_t)r * N + 16] = c01[r];
    Cp[((size_t)r + 16) * N] = c10[r];
    Cp[((size_t)r + 16) * N + 16] = c11[r];
  }
}

// ======  RMSNorm + RoPE + scatter, emitting pre-split bf16 operands  =======
// One wave per (b, s, slot); slot 0..15 = q-head (scaled by 1/sqrt(HD)),
// 16..19 = k-head, 20..23 = v-head (split + transpose to [b,kv,hd,s]).
__global__ __launch_bounds__(256) void norm_rope_scatter(
    const float* __restrict__ qkv, const float* __restrict__ fc,
    const float* __restrict__ qw, const float* __restrict__ kw,
    __bf16* __restrict__ qhi, __bf16* __restrict__ qlo,
    __bf16* __restrict__ khi, __bf16* __restrict__ klo,
    __bf16* __restrict__ vthi, __bf16* __restrict__ vtlo) {
  const int lane = threadIdx.x & 31;
  const int wave = threadIdx.x >> 5;
  const int flat = blockIdx.x * 8 + wave;
  const int b = flat / (S_LEN * 24);
  int rem = flat - b * (S_LEN * 24);
  const int s = rem / 24;
  const int slot = rem - s * 24;

  const float* base = qkv + (size_t)(b * S_LEN + s) * 3072;
  const float* src;
  if (slot < 16)      src = base + slot * HDIM;
  else if (slot < 20) src = base + 2048 + (slot - 16) * HDIM;
  else                src = base + 2560 + (slot - 20) * HDIM;

  const float4 xv = ((const float4*)src)[lane];

  if (slot >= 20) {  // v: split + transpose to [b, kv, hd, s]
    const size_t rbase = ((size_t)(b * NKVH + (slot - 20)) * HDIM + 4 * lane) * S_LEN + s;
    float vv[4] = {xv.x, xv.y, xv.z, xv.w};
#pragma unroll
    for (int c = 0; c < 4; c++) {
      const bfpair p = split1(vv[c]);
      vthi[rbase + (size_t)c * S_LEN] = p.h;
      vtlo[rbase + (size_t)c * S_LEN] = p.l;
    }
    return;
  }

  float ss = xv.x * xv.x + xv.y * xv.y + xv.z * xv.z + xv.w * xv.w;
#pragma unroll
  for (int off = 16; off > 0; off >>= 1) ss += __shfl_xor(ss, off, 32);
  float inv = rsqrtf(ss * (1.0f / 128.0f) + 1e-6f);
  if (slot < 16) inv *= 0.088388347648318447f;  // fold 1/sqrt(HD) into q

  const float* w = (slot < 16) ? qw : kw;
  const float4 wv = ((const float4*)w)[lane];
  const float a0 = xv.x * inv * wv.x, a1 = xv.y * inv * wv.y;
  const float a2 = xv.z * inv * wv.z, a3 = xv.w * inv * wv.w;

  // freqs_cis[s, i, c] flat = s*128 + 2*i; this lane owns pairs i=2*lane, 2*lane+1
  const float2* f = (const float2*)(fc + (size_t)s * 128);
  const float2 f0 = f[2 * lane];
  const float2 f1 = f[2 * lane + 1];
  const float o0 = a0 * f0.x - a1 * f0.y, o1 = a1 * f0.x + a0 * f0.y;
  const float o2 = a2 * f1.x - a3 * f1.y, o3 = a3 * f1.x + a2 * f1.y;

  v4bf h, l;
  bfpair p;
  p = split1(o0); h[0] = p.h; l[0] = p.l;
  p = split1(o1); h[1] = p.h; l[1] = p.l;
  p = split1(o2); h[2] = p.h; l[2] = p.l;
  p = split1(o3); h[3] = p.h; l[3] = p.l;

  size_t dbase;
  __bf16 *dh, *dl;
  if (slot < 16) {
    dbase = (((size_t)(b * NHEADS + slot)) * S_LEN + s) * HDIM;
    dh = qhi; dl = qlo;
  } else {
    dbase = (((size_t)(b * NKVH + (slot - 16))) * S_LEN + s) * HDIM;
    dh = khi; dl = klo;
  }
  ((v4bf*)(dh + dbase))[lane] = h;
  ((v4bf*)(dl + dbase))[lane] = l;
}

// =====================  Causal flash attention  ===========================
// One wave (workgroup of 32) per 16-row q-tile of one (b, h).  Scores and
// P.V both bf16x3 on v_wmma_f32_16x16x32_bf16.  P transposed C->A layout
// through 2 KB LDS guarded by s_wait_dscnt.  All operands pre-split bf16.
__global__ __launch_bounds__(32) void flash_attn_kernel(
    const __bf16* __restrict__ qhi, const __bf16* __restrict__ qlo,
    const __bf16* __restrict__ khi, const __bf16* __restrict__ klo,
    const __bf16* __restrict__ vthi, const __bf16* __restrict__ vtlo,
    float* __restrict__ y) {
  __shared__ __bf16 pbh[16 * 32];
  __shared__ __bf16 pbl[16 * 32];

  const int lane = threadIdx.x;
  const int nqt = S_LEN / 16;
  const int qt = blockIdx.x % nqt;
  const int bh = blockIdx.x / nqt;
  const int h = bh % NHEADS;
  const int b = bh / NHEADS;
  const int kv = h / (NHEADS / NKVH);
  const int q0 = qt * 16;

  const size_t qoff = (((size_t)b * NHEADS + h) * S_LEN + q0) * HDIM;
  const size_t koff = ((size_t)b * NKVH + kv) * S_LEN * HDIM;
  const size_t voff = ((size_t)b * NKVH + kv) * (size_t)HDIM * S_LEN;  // [hd][s]

  const int cn = lane & 15;          // C-layout column / frag row selector
  const int cg = (lane >> 4) * 8;    // C-layout row group
  const int g16 = (lane >> 4) * 16;  // B-frag K group

  // Q fragments (scale already folded in by norm_rope_scatter)
  v16bf qfh[4], qfl[4];
  const size_t qrow = qoff + (size_t)cn * HDIM;
#pragma unroll
  for (int st = 0; st < 4; ++st) {
    qfh[st] = load_a_frag(qhi + qrow + st * 32, lane);
    qfl[st] = load_a_frag(qlo + qrow + st * 32, lane);
  }

  v8f o[8];
#pragma unroll
  for (int t = 0; t < 8; t++)
#pragma unroll
    for (int r = 0; r < 8; r++) o[t][r] = 0.0f;

  float mrow[8], lrow[8];
#pragma unroll
  for (int r = 0; r < 8; r++) { mrow[r] = -3.0e38f; lrow[r] = 0.0f; }

  const int nk = q0 + 16;  // causal: keys [0, q0+16); j <= 2016 so no OOB
  for (int j = 0; j < nk; j += 32) {
    // ---- scores S = Qs @ K^T, bf16x3, two 16-wide key subtiles ----
    const size_t k0r = koff + (size_t)(j + cn) * HDIM;
    const size_t k1r = koff + (size_t)(j + 16 + cn) * HDIM;
    v8f s0, s1;
#pragma unroll
    for (int r = 0; r < 8; r++) { s0[r] = 0.0f; s1[r] = 0.0f; }
#pragma unroll
    for (int st = 0; st < 4; ++st) {
      const v16bf k0h = load_b_frag(khi + k0r + st * 32, lane);
      const v16bf k0l = load_b_frag(klo + k0r + st * 32, lane);
      const v16bf k1h = load_b_frag(khi + k1r + st * 32, lane);
      const v16bf k1l = load_b_frag(klo + k1r + st * 32, lane);
      s0 = wmma_bf16(qfh[st], k0h, s0);
      s0 = wmma_bf16(qfl[st], k0h, s0);
      s0 = wmma_bf16(qfh[st], k0l, s0);
      s1 = wmma_bf16(qfh[st], k1h, s1);
      s1 = wmma_bf16(qfl[st], k1h, s1);
      s1 = wmma_bf16(qfh[st], k1l, s1);
    }

    // ---- online softmax (per C-layout row, 16-lane shuffle reductions) ----
    float alpha[8];
#pragma unroll
    for (int r = 0; r < 8; r++) {
      const int qr = q0 + cg + r;
      const bool ok0 = (j + cn) <= qr;
      const bool ok1 = (j + 16 + cn) <= qr;
      float v0 = ok0 ? s0[r] : -3.0e38f;
      float v1 = ok1 ? s1[r] : -3.0e38f;
      float mx = fmaxf(v0, v1);
#pragma unroll
      for (int off = 1; off < 16; off <<= 1) mx = fmaxf(mx, __shfl_xor(mx, off, 32));
      const float mnew = fmaxf(mrow[r], mx);
      const float p0 = ok0 ? __expf(s0[r] - mnew) : 0.0f;
      const float p1 = ok1 ? __expf(s1[r] - mnew) : 0.0f;
      float rs = p0 + p1;
#pragma unroll
      for (int off = 1; off < 16; off <<= 1) rs += __shfl_xor(rs, off, 32);
      alpha[r] = __expf(mrow[r] - mnew);
      lrow[r] = lrow[r] * alpha[r] + rs;
      mrow[r] = mnew;
      const bfpair pp0 = split1(p0);
      const bfpair pp1 = split1(p1);
      pbh[(cg + r) * 32 + cn] = pp0.h;
      pbl[(cg + r) * 32 + cn] = pp0.l;
      pbh[(cg + r) * 32 + 16 + cn] = pp1.h;
      pbl[(cg + r) * 32 + 16 + cn] = pp1.l;
    }

    // rescale running O
#pragma unroll
    for (int t = 0; t < 8; t++)
#pragma unroll
      for (int r = 0; r < 8; r++) o[t][r] *= alpha[r];

    // P: C layout -> A-fragment via LDS (same-wave; dscnt fence + compiler fence)
    asm volatile("s_wait_dscnt 0" ::: "memory");
    v16bf pah, pal;
    {
      const int prow = lane & 15;
      const int pg = (lane >> 4) * 16;
#pragma unroll
      for (int i = 0; i < 8; i++) {
        pah[i] = pbh[prow * 32 + pg + i];
        pah[8 + i] = pbh[prow * 32 + pg + 8 + i];
        pal[i] = pbl[prow * 32 + pg + i];
        pal[8 + i] = pbl[prow * 32 + pg + 8 + i];
      }
    }

    // ---- O += P @ V, bf16x3, V pre-split + pre-transposed [hd][s] ----
#pragma unroll
    for (int t = 0; t < 8; t++) {
      const size_t vrow = voff + (size_t)(t * 16 + cn) * S_LEN + j + g16;
      const v16bf vfh = cat8(ld8(vthi + vrow), ld8(vthi + vrow + 8));
      const v16bf vfl = cat8(ld8(vtlo + vrow), ld8(vtlo + vrow + 8));
      o[t] = wmma_bf16(pah, vfh, o[t]);
      o[t] = wmma_bf16(pal, vfh, o[t]);
      o[t] = wmma_bf16(pah, vfl, o[t]);
    }
    asm volatile("" ::: "memory");
  }

  // ---- normalize and store y[b, s, h, hd] (fp32) ----
#pragma unroll
  for (int r = 0; r < 8; r++) {
    const float inv = 1.0f / lrow[r];
    float* dst = y + ((size_t)b * S_LEN + q0 + cg + r) * (NHEADS * HDIM) +
                 (size_t)h * HDIM + cn;
#pragma unroll
    for (int t = 0; t < 8; t++) dst[t * 16] = o[t][r] * inv;
  }
}

// ============================  launcher  ==================================
extern "C" void kernel_launch(void* const* d_in, const int* in_sizes, int n_in,
                              void* d_out, int out_size, void* d_ws, size_t ws_size,
                              hipStream_t stream) {
  (void)in_sizes; (void)n_in; (void)out_size; (void)ws_size;
  const float* x    = (const float*)d_in[0];
  const float* fc   = (const float*)d_in[1];
  const float* wqkv = (const float*)d_in[2];
  const float* wo   = (const float*)d_in[3];
  const float* qw   = (const float*)d_in[4];
  const float* kw   = (const float*)d_in[5];
  float* out = (float*)d_out;
  char* ws = (char*)d_ws;

  // Workspace carve (bytes, all 256-aligned).  Total ~152 MiB.
  //   [0, 50331648)              qkv f32 [4096,3072]; y f32 [4096,2048] aliases it
  //   xhi/xlo (8.39M bf16 each)  -> reused as yhi/ylo after flash
  //   whi/wlo (6.29M bf16 each)  -> wqkv splits, reused for wo splits
  //   qhi/qlo (8.39M), khi/klo (2.10M), vthi/vtlo (2.10M)
  float* qkv = (float*)ws;
  float* yf  = (float*)ws;  // alias: qkv dead after norm_rope_scatter
  size_t off = 50331648;
  __bf16* xhi = (__bf16*)(ws + off); off += 16777216;
  __bf16* xlo = (__bf16*)(ws + off); off += 16777216;
  __bf16* whi = (__bf16*)(ws + off); off += 12582912;
  __bf16* wlo = (__bf16*)(ws + off); off += 12582912;
  __bf16* qhi = (__bf16*)(ws + off); off += 16777216;
  __bf16* qlo = (__bf16*)(ws + off); off += 16777216;
  __bf16* khi = (__bf16*)(ws + off); off += 4194304;
  __bf16* klo = (__bf16*)(ws + off); off += 4194304;
  __bf16* vthi = (__bf16*)(ws + off); off += 4194304;
  __bf16* vtlo = (__bf16*)(ws + off); off += 4194304;
  __bf16* yhi = xhi;  // x splits dead after QKV GEMM
  __bf16* ylo = xlo;

  // 1) one-shot bf16x3 splits of x and wqkv
  split_f32_kernel<<<(8388608 / 4 + 255) / 256, 256, 0, stream>>>(x, xhi, xlo, 8388608 / 4);
  split_f32_kernel<<<(6291456 / 4 + 255) / 256, 256, 0, stream>>>(wqkv, whi, wlo, 6291456 / 4);

  // 2) qkv = x @ wqkv^T   (M=4096, N=3072, K=2048)
  gemm_nt_bf16x3<<<dim3(3072 / 128, 4096 / 64), dim3(256), 0, stream>>>(
      xhi, xlo, whi, wlo, qkv, 4096, 3072, 2048);

  // 3) RMSNorm + RoPE, emit pre-split q/k and pre-split transposed v
  norm_rope_scatter<<<(2 * S_LEN * 24) / 8, 256, 0, stream>>>(
      qkv, fc, qw, kw, qhi, qlo, khi, klo, vthi, vtlo);

  // 4) causal GQA flash attention -> y f32 [b, s, h, hd]
  flash_attn_kernel<<<2 * NHEADS * (S_LEN / 16), 32, 0, stream>>>(
      qhi, qlo, khi, klo, vthi, vtlo, yf);

  // 5) splits of y and wo (reusing x/wqkv split regions)
  split_f32_kernel<<<(8388608 / 4 + 255) / 256, 256, 0, stream>>>(yf, yhi, ylo, 8388608 / 4);
  split_f32_kernel<<<(4194304 / 4 + 255) / 256, 256, 0, stream>>>(wo, whi, wlo, 4194304 / 4);

  // 6) out = y @ wo^T   (M=4096, N=2048, K=2048)
  gemm_nt_bf16x3<<<dim3(2048 / 128, 4096 / 64), dim3(256), 0, stream>>>(
      yhi, ylo, whi, wlo, out, 4096, 2048, 2048);
}
